// ResidualLinear4bit_9517647528630
// MI455X (gfx1250) — matrix operand order repaired and checked by reading, not compiled, not run on hardware
//
#include <hip/hip_runtime.h>
#include <hip/hip_bf16.h>

typedef __attribute__((ext_vector_type(16))) _Float16 v16h;
typedef __attribute__((ext_vector_type(8)))  _Float16 v8h;
typedef __attribute__((ext_vector_type(8)))  float    v8f;
typedef __attribute__((ext_vector_type(4)))  float    v4f;
typedef __attribute__((ext_vector_type(4)))  int      v4i;

#define K_DIM    4096
#define BLK_M    256
#define BLK_N    128
#define BLK_K    32
#define NTHREADS 512
#define LDS_H    40   // halves per LDS row: 80B -> 20 dwords, conflict-free 16B column reads

// NF4 codebook (bitsandbytes)
__device__ __constant__ float NF4_TAB[16] = {
    -1.0f, -0.6961928009986877f, -0.5250730514526367f, -0.39491748809814453f,
    -0.28444138169288635f, -0.18477343022823334f, -0.09105003625154495f, 0.0f,
    0.07958029955625534f, 0.16093020141124725f, 0.24611230194568634f,
    0.33791524171829224f, 0.44070982933044434f, 0.5626170039176941f,
    0.7229568362236023f, 1.0f};

__global__ __launch_bounds__(NTHREADS)
void nf4_linear_wmma_kernel(const float* __restrict__ x,
                            const int*   __restrict__ w_idx,
                            const float* __restrict__ absmax,
                            const float* __restrict__ bias,
                            float*       __restrict__ out,
                            int Nout)
{
    // Double-buffered stages: 2*(256+128)*40 halves = 61,440 B (< 64 KB static)
    __shared__ __align__(16) _Float16 sA[2][BLK_M][LDS_H];   // x tile (f16)
    __shared__ __align__(16) _Float16 sB[2][BLK_N][LDS_H];   // dequantized W tile (f16)

    const int tid  = threadIdx.x;
    const int lane = tid & 31;
    const int wave = tid >> 5;            // 0..15
    const int l15  = lane & 15;
    const int hi   = lane >> 4;           // 0 for lanes 0-15, 1 for lanes 16-31

    const int wm = wave & 3;              // 4 M-groups of 64 rows (4 subtiles each)
    const int wn = wave >> 2;             // 4 N-groups of 32 cols (2 subtiles each)

    const int m_blk = blockIdx.x * BLK_M; // M fastest-varying -> L2 reuse of w columns
    const int n_blk = blockIdx.y * BLK_N;

    // each lane caches one codebook entry; ds_bpermute does the 16-way LUT
    const int code_i = __float_as_int(NF4_TAB[l15]);

    // ---- staging assignments (512 threads) ----
    // x tile: 256 rows x 32 cols f32 -> 2 threads per row, 16 floats each
    const int xRow = tid >> 1;            // 0..255
    const int xCol = (tid & 1) * 16;
    // w tile: 128 rows x 32 cols i32 -> 4 threads per row, 8 ints each
    const int wRow = tid >> 2;            // 0..127
    const int wCol = (tid & 3) * 8;

    const float* xPtr = x     + (size_t)(m_blk + xRow) * K_DIM + xCol;
    const int*   wPtr = w_idx + (size_t)(n_blk + wRow) * K_DIM + wCol;
    const float* aPtr = absmax + (size_t)(n_blk + wRow) * (K_DIM / 64);

    // two register staging sets -> global prefetch 2 stages (64 K-elems) ahead
    v4f   xs[2][4];
    v4i   ws[2][2];
    float scale[2];

    auto load_stage = [&](int s, int k0) {
        const v4f* xp = (const v4f*)(xPtr + k0);
        const v4i* wp = (const v4i*)(wPtr + k0);
#pragma unroll
        for (int i = 0; i < 4; ++i) xs[s][i] = xp[i];
#pragma unroll
        for (int i = 0; i < 2; ++i) ws[s][i] = wp[i];
        scale[s] = aPtr[k0 >> 6];         // NF4 blocksize 64 spans two 32-K stages
    };

    auto store_stage = [&](int buf, int s) {
        // x -> f16 -> sA
        v8h ha[2];
#pragma unroll
        for (int i = 0; i < 4; ++i)
#pragma unroll
            for (int j = 0; j < 4; ++j) {
                const int e = i * 4 + j;
                ha[e >> 3][e & 7] = (_Float16)xs[s][i][j];
            }
        v8h* dA = (v8h*)&sA[buf][xRow][xCol];
        dA[0] = ha[0];
        dA[1] = ha[1];

        // w_idx -> codebook gather (ds_bpermute) -> * absmax -> f16 -> sB
        v8h hb;
#pragma unroll
        for (int i = 0; i < 2; ++i)
#pragma unroll
            for (int j = 0; j < 4; ++j) {
                const float c = __int_as_float(
                    __builtin_amdgcn_ds_bpermute(ws[s][i][j] << 2, code_i));
                hb[i * 4 + j] = (_Float16)(c * scale[s]);
            }
        *(v8h*)&sB[buf][wRow][wCol] = hb;
    };

    v8f acc[4][2] = {};

    auto compute = [&](int buf) {
        v16h a[4], b[2];
        // A fragment: 16x32 f16, lane = M row; halves 0-7 -> K=koff..+7,
        // halves 8-15 -> K=koff+16..+23, koff = 8*hi  (ISA 16-bit A layout)
#pragma unroll
        for (int ms = 0; ms < 4; ++ms) {
            const _Float16* pa = &sA[buf][wm * 64 + ms * 16 + l15][hi * 8];
            const v8h lo = *(const v8h*)pa;
            const v8h hh = *(const v8h*)(pa + 16);
#pragma unroll
            for (int j = 0; j < 8; ++j) { a[ms][j] = lo[j]; a[ms][j + 8] = hh[j]; }
        }
        // B fragment: 32x16 f16, lane = N col; halves 0-7 -> K=koff..+7, halves 8-15 -> K=koff+16..+23, koff = 8*hi
#pragma unroll
        for (int ns = 0; ns < 2; ++ns) {
            const _Float16* pb = &sB[buf][wn * 32 + ns * 16 + l15][hi * 8];
            const v8h lo = *(const v8h*)pb;
            const v8h hh = *(const v8h*)(pb + 16);
#pragma unroll
            for (int j = 0; j < 8; ++j) { b[ns][j] = lo[j]; b[ns][j + 8] = hh[j]; }
        }
#pragma unroll
        for (int ms = 0; ms < 4; ++ms)
#pragma unroll
            for (int ns = 0; ns < 2; ++ns)
                acc[ms][ns] = __builtin_amdgcn_wmma_f32_16x16x32_f16(
                    false, a[ms], false, b[ns], (short)0, acc[ms][ns],
                    false, false);
    };

    load_stage(0, 0);
    load_stage(1, BLK_K);
    for (int k0 = 0; k0 < K_DIM; k0 += 2 * BLK_K) {
        // even stage -> buffer 0
        store_stage(0, 0);
        __syncthreads();
        if (k0 + 2 * BLK_K < K_DIM) load_stage(0, k0 + 2 * BLK_K);
        compute(0);
        // odd stage -> buffer 1 (overlaps with stragglers computing buffer 0)
        store_stage(1, 1);
        __syncthreads();
        if (k0 + 3 * BLK_K < K_DIM) load_stage(1, k0 + 3 * BLK_K);
        compute(1);
    }

    // Epilogue: C/D layout — VGPR r: lanes 0-15 M=r, lanes 16-31 M=8+r; N = lane&15
    float bval[2];
#pragma unroll
    for (int ns = 0; ns < 2; ++ns)
        bval[ns] = bias[n_blk + wn * 32 + ns * 16 + l15];

#pragma unroll
    for (int ms = 0; ms < 4; ++ms)
#pragma unroll
        for (int ns = 0; ns < 2; ++ns) {
            const int mg = m_blk + wm * 64 + ms * 16 + hi * 8;
            const int ng = n_blk + wn * 32 + ns * 16 + l15;
            float* op = out + (size_t)mg * Nout + ng;
#pragma unroll
            for (int r = 0; r < 8; ++r)
                op[(size_t)r * Nout] = acc[ms][ns][r] + bval[ns];
        }
}

extern "C" void kernel_launch(void* const* d_in, const int* in_sizes, int n_in,
                              void* d_out, int out_size, void* d_ws, size_t ws_size,
                              hipStream_t stream) {
    (void)n_in; (void)out_size; (void)d_ws; (void)ws_size;
    const float* x      = (const float*)d_in[0];
    const int*   w_idx  = (const int*)d_in[1];
    const float* absmax = (const float*)d_in[2];
    const float* bias   = (const float*)d_in[3];
    float*       out    = (float*)d_out;

    const int Nout = in_sizes[3];           // 11008
    const int Mtot = in_sizes[0] / K_DIM;   // 2*512 = 1024

    dim3 grid(Mtot / BLK_M, Nout / BLK_N);  // (4, 86); M fastest -> w reuse in L2
    nf4_linear_wmma_kernel<<<grid, NTHREADS, 0, stream>>>(x, w_idx, absmax, bias, out, Nout);
}
